// Net_36859409334357
// MI455X (gfx1250) — matrix operand order repaired and checked by reading, not compiled, not run on hardware
//
#include <hip/hip_runtime.h>
#include <hip/hip_bf16.h>

// ---------------- problem constants (match reference) ----------------
#define BG   128            // graphs
#define NG   1024           // nodes per graph
#define DEGG 16
#define EPG  (NG * DEGG)    // 16384 edges per graph (contiguous per graph)
#define BN   (BG * NG)      // 131072 nodes
#define ETOT (BN * DEGG)    // 2097152 edges
#define HF   64             // hidden features

typedef __attribute__((ext_vector_type(16))) _Float16 v16h;
typedef __attribute__((ext_vector_type(8)))  _Float16 v8h;
typedef __attribute__((ext_vector_type(8)))  float    v8f;

// =====================================================================
// conv1 linear: h = x @ W1   (x:[BN,2], W1:[2,64])  -- K=2, VALU is fine
// =====================================================================
__global__ void lin1_kernel(const float* __restrict__ x,
                            const float* __restrict__ W1,
                            float* __restrict__ out) {
    int idx = blockIdx.x * 256 + threadIdx.x;
    if (idx >= BN * HF) return;
    int n = idx >> 6, f = idx & 63;
    out[idx] = x[2 * n] * W1[f] + x[2 * n + 1] * W1[64 + f];
}

// =====================================================================
// WMMA GEMM: out[BN,64] = hin[BN,64] @ W[64,64]   (f16 in, f32 acc)
// Block = 256 threads = 8 waves; each wave owns a 16-row tile.
// W is staged transposed (Wt[col][k]) in LDS as f16.
// =====================================================================
__device__ __forceinline__ v16h load_a_frag(const float* __restrict__ hrow,
                                            int k0, int lane) {
    // 16-bit A 16x32 layout: lanes 0-15 K=k0+{0..7,16..23}, lanes 16-31 +8
    int kb = k0 + ((lane & 16) ? 8 : 0);
    const float4* p0 = (const float4*)(hrow + kb);
    const float4* p1 = (const float4*)(hrow + kb + 16);
    float4 x0 = p0[0], x1 = p0[1], x2 = p1[0], x3 = p1[1];
    v16h a;
    a[0]  = (_Float16)x0.x; a[1]  = (_Float16)x0.y;
    a[2]  = (_Float16)x0.z; a[3]  = (_Float16)x0.w;
    a[4]  = (_Float16)x1.x; a[5]  = (_Float16)x1.y;
    a[6]  = (_Float16)x1.z; a[7]  = (_Float16)x1.w;
    a[8]  = (_Float16)x2.x; a[9]  = (_Float16)x2.y;
    a[10] = (_Float16)x2.z; a[11] = (_Float16)x2.w;
    a[12] = (_Float16)x3.x; a[13] = (_Float16)x3.y;
    a[14] = (_Float16)x3.z; a[15] = (_Float16)x3.w;
    return a;
}

__device__ __forceinline__ v16h load_b_frag(const _Float16* __restrict__ Wt,
                                            int n0, int k0, int lane) {
    // 16-bit B 32x16 layout: lane = column, lanes 0-15 K=k0+{0..7,16..23}, lanes 16-31 +8
    int c  = n0 + (lane & 15);
    int kb = k0 + ((lane & 16) ? 8 : 0);
    const v8h* p = (const v8h*)(Wt + c * 64 + kb);   // 16B-aligned
    v8h lo = p[0], hi = p[2];
    v16h b;
#pragma unroll
    for (int i = 0; i < 8; ++i) { b[i] = lo[i]; b[8 + i] = hi[i]; }
    return b;
}

__global__ void gemm_h64_wmma(const float* __restrict__ hin,
                              const float* __restrict__ W,   // [64][64] row-major f32
                              float* __restrict__ out) {
    __shared__ _Float16 Wt[64 * 64];                  // transposed, f16 (8 KB)
    int tid = threadIdx.x;
    for (int i = tid; i < 4096; i += 256) {
        int k = i >> 6, c = i & 63;
        Wt[c * 64 + k] = (_Float16)W[k * 64 + c];
    }
    __syncthreads();

    int wave = tid >> 5, lane = tid & 31;
    int m0 = blockIdx.x * 128 + wave * 16;            // 16-row tile per wave
    const float* hrow = hin + (size_t)(m0 + (lane & 15)) * 64;

    v16h a0 = load_a_frag(hrow, 0,  lane);
    v16h a1 = load_a_frag(hrow, 32, lane);

#pragma unroll
    for (int ct = 0; ct < 4; ++ct) {                  // 4 column tiles of 16
        v8f acc = {};
        v16h b0 = load_b_frag(Wt, ct * 16, 0,  lane);
        v16h b1 = load_b_frag(Wt, ct * 16, 32, lane);
        acc = __builtin_amdgcn_wmma_f32_16x16x32_f16(false, a0, false, b0,
                                                     (short)0, acc, false, false);
        acc = __builtin_amdgcn_wmma_f32_16x16x32_f16(false, a1, false, b1,
                                                     (short)0, acc, false, false);
        // D 16x16 f32 layout: lane = col, VGPR r = row (+8 for lanes 16-31)
        int cc = ct * 16 + (lane & 15);
        int rb = m0 + ((lane & 16) ? 8 : 0);
#pragma unroll
        for (int r = 0; r < 8; ++r)
            out[(size_t)(rb + r) * 64 + cc] = acc[r];
    }
}

// =====================================================================
// GCN aggregation over 64 features, LDS-resident per (graph, half):
//   deg -> rsqrt -> scatter norm*hW[src] (ds_add_f32) -> +self +bias (+relu)
// grid = (BG, 2); dynamic LDS = 1024*32 acc + 1024 dinv floats (132 KB)
// =====================================================================
__global__ void gcn_agg_feat(const float* __restrict__ hW,
                             const int*   __restrict__ src,
                             const int*   __restrict__ dst,
                             const float* __restrict__ eattr,
                             const int*   __restrict__ alive,   // null = all alive
                             int use_attr,
                             const float* __restrict__ bias,
                             int do_relu,
                             float* __restrict__ out) {
    extern __shared__ float smem[];
    float* acc  = smem;                    // [1024][32]
    float* dinv = smem + 1024 * 32;        // [1024] (deg, then rsqrt)
    int g = blockIdx.x, fh = blockIdx.y, tid = threadIdx.x;
    int nbase = g * NG, ebase = g * EPG;

    for (int i = tid; i < 1024 * 32; i += 256) acc[i] = 0.f;
    for (int i = tid; i < 1024; i += 256) dinv[i] = 1.0f;   // self-loop weight
    __syncthreads();

    // degree pass (also prefetch the rows the scatter pass will gather)
    for (int e = tid; e < EPG; e += 256) {
        int s = src[ebase + e], d = dst[ebase + e];
        float w = use_attr ? eattr[ebase + e] : 1.0f;
        if (alive && !(alive[s] && alive[d])) w = 0.f;
        if (w != 0.f) {
            atomicAdd(&dinv[d - nbase], w);
            __builtin_prefetch(hW + (size_t)s * 64 + fh * 32, 0, 0);
        }
    }
    __syncthreads();
    for (int i = tid; i < 1024; i += 256) dinv[i] = rsqrtf(dinv[i]);
    __syncthreads();

    // scatter pass: 8 threads per edge, float4 per thread
    for (int e0 = 0; e0 < EPG; e0 += 32) {
        int e = ebase + e0 + (tid >> 3);
        int s = src[e], d = dst[e];
        float w = use_attr ? eattr[e] : 1.0f;
        if (alive && !(alive[s] && alive[d])) w = 0.f;
        if (w != 0.f) {
            int sl = s - nbase, dl = d - nbase;
            float nrm = dinv[sl] * w * dinv[dl];
            int j4 = (tid & 7) * 4;
            float4 hv = *(const float4*)(hW + (size_t)s * 64 + fh * 32 + j4);
            float* ap = acc + dl * 32 + j4;
            atomicAdd(ap + 0, nrm * hv.x);
            atomicAdd(ap + 1, nrm * hv.y);
            atomicAdd(ap + 2, nrm * hv.z);
            atomicAdd(ap + 3, nrm * hv.w);
        }
    }
    __syncthreads();

    for (int i = tid; i < 1024 * 32; i += 256) {
        int n = i >> 5, j = i & 31, f = fh * 32 + j;
        float di = dinv[n];
        float v = acc[i] + di * di * hW[(size_t)(nbase + n) * 64 + f] + bias[f];
        if (do_relu) v = fmaxf(v, 0.f);
        out[(size_t)(nbase + n) * 64 + f] = v;
    }
}

// =====================================================================
// SAGPool score GCN (H->1), per-graph, fully LDS-resident (12 KB)
// =====================================================================
__global__ void gcn_score(const float* __restrict__ h,
                          const int*   __restrict__ src,
                          const int*   __restrict__ dst,
                          const int*   __restrict__ alive,   // null = all alive
                          const float* __restrict__ Wp,      // [64]
                          const float* __restrict__ bp,      // [1]
                          float* __restrict__ score) {
    __shared__ float hwp[1024];
    __shared__ float dinv[1024];
    __shared__ float acc[1024];
    __shared__ float wpl[64];
    int g = blockIdx.x, tid = threadIdx.x;
    int nbase = g * NG, ebase = g * EPG;

    if (tid < 64) wpl[tid] = Wp[tid];
    for (int i = tid; i < 1024; i += 256) { dinv[i] = 1.f; acc[i] = 0.f; }
    __syncthreads();

    for (int n = tid; n < 1024; n += 256) {
        const float* hr = h + (size_t)(nbase + n) * 64;
        float s = 0.f;
#pragma unroll
        for (int k = 0; k < 64; ++k) s += hr[k] * wpl[k];
        hwp[n] = s;
    }
    __syncthreads();

    for (int e = tid; e < EPG; e += 256) {
        int s = src[ebase + e], d = dst[ebase + e];
        float w = 1.f;
        if (alive && !(alive[s] && alive[d])) w = 0.f;
        if (w != 0.f) atomicAdd(&dinv[d - nbase], w);
    }
    __syncthreads();
    for (int i = tid; i < 1024; i += 256) dinv[i] = rsqrtf(dinv[i]);
    __syncthreads();

    for (int e = tid; e < EPG; e += 256) {
        int s = src[ebase + e], d = dst[ebase + e];
        float w = 1.f;
        if (alive && !(alive[s] && alive[d])) w = 0.f;
        if (w != 0.f) {
            int sl = s - nbase, dl = d - nbase;
            atomicAdd(&acc[dl], dinv[sl] * w * dinv[dl] * hwp[sl]);
        }
    }
    __syncthreads();

    float bpv = bp[0];
    for (int n = tid; n < 1024; n += 256) {
        float di = dinv[n];
        score[nbase + n] = acc[n] + di * di * hwp[n] + bpv;
    }
}

// =====================================================================
// SAGPool top-k: in-LDS bitonic sort (descending) of masked scores,
// threshold + smallest-index tie break (matches lax.top_k), then
// xn = keep ? h * tanh(score) : 0 and alive_new.
// =====================================================================
__global__ void sag_topk(const float* __restrict__ score,
                         const int*   __restrict__ alive,   // null = all alive
                         const float* __restrict__ h,
                         int K,
                         int*   __restrict__ alive_new,
                         float* __restrict__ xn) {
    __shared__ float S[1024];
    __shared__ float tn[1024];
    __shared__ unsigned char keep[1024];
    __shared__ int cnt_gt;
    __shared__ float thrS;
    int g = blockIdx.x, tid = threadIdx.x;
    int nbase = g * NG;
    const float NEG = -__builtin_inff();

    for (int n = tid; n < 1024; n += 256) {
        float s = score[nbase + n];
        S[n] = (alive == nullptr || alive[nbase + n]) ? s : NEG;
    }
    if (tid == 0) cnt_gt = 0;
    __syncthreads();

    // bitonic sort, descending
    for (int kk = 2; kk <= 1024; kk <<= 1) {
        for (int j = kk >> 1; j > 0; j >>= 1) {
            for (int i = tid; i < 1024; i += 256) {
                int p = i ^ j;
                if (p > i) {
                    bool up = ((i & kk) == 0);
                    float a = S[i], b = S[p];
                    bool sw = up ? (a < b) : (a > b);
                    if (sw) { S[i] = b; S[p] = a; }
                }
            }
            __syncthreads();
        }
    }
    if (tid == 0) thrS = S[K - 1];
    __syncthreads();
    float thr = thrS;

    int lc = 0;
    for (int i = tid; i < 1024; i += 256) if (S[i] > thr) lc++;
    if (lc) atomicAdd(&cnt_gt, lc);
    __syncthreads();
    int need_eq = K - cnt_gt;

    for (int n = tid; n < 1024; n += 256) {
        float sraw = score[nbase + n];
        float s = (alive == nullptr || alive[nbase + n]) ? sraw : NEG;
        bool kp = false;
        if (s > thr) kp = true;
        else if (s == thr) {              // rare tie path: stable by index
            int rank = 0;
            for (int j = 0; j < n; ++j) {
                float sj = score[nbase + j];
                sj = (alive == nullptr || alive[nbase + j]) ? sj : NEG;
                if (sj == thr) rank++;
            }
            kp = (rank < need_eq);
        }
        keep[n] = kp ? 1 : 0;
        tn[n] = tanhf(sraw);
        alive_new[nbase + n] = kp ? 1 : 0;
    }
    __syncthreads();

    for (int i = tid; i < 1024 * 64; i += 256) {
        int n = i >> 6;
        xn[(size_t)nbase * 64 + i] =
            keep[n] ? h[(size_t)nbase * 64 + i] * tn[n] : 0.f;
    }
}

// =====================================================================
// readout: concat(max, mean) over alive nodes; accumulate into gacc
// =====================================================================
__global__ void readout_kernel(const float* __restrict__ xn,
                               const int*   __restrict__ alive,
                               float* __restrict__ gacc,
                               int accumulate) {
    __shared__ float pmax[4][64];
    __shared__ float psum[4][64];
    __shared__ int cnt;
    int g = blockIdx.x, tid = threadIdx.x;
    int f = tid & 63, c = tid >> 6;
    int nbase = g * NG;
    if (tid == 0) cnt = 0;
    __syncthreads();

    float mx = -__builtin_inff(), sm = 0.f;
    int lc = 0;
    for (int n = c; n < 1024; n += 4) {
        if (alive[nbase + n]) {
            float v = xn[(size_t)(nbase + n) * 64 + f];
            mx = fmaxf(mx, v);
            sm += v;
            if (f == 0) lc++;
        }
    }
    pmax[c][f] = mx; psum[c][f] = sm;
    if (f == 0 && lc) atomicAdd(&cnt, lc);
    __syncthreads();

    if (tid < 64) {
        float m2 = fmaxf(fmaxf(pmax[0][tid], pmax[1][tid]),
                         fmaxf(pmax[2][tid], pmax[3][tid]));
        float s2 = psum[0][tid] + psum[1][tid] + psum[2][tid] + psum[3][tid];
        float mean = s2 / (float)cnt;
        if (accumulate) {
            gacc[g * 128 + tid]      += m2;
            gacc[g * 128 + 64 + tid] += mean;
        } else {
            gacc[g * 128 + tid]      = m2;
            gacc[g * 128 + 64 + tid] = mean;
        }
    }
}

// =====================================================================
// final MLP + log_softmax, one workgroup per graph row
// =====================================================================
__global__ void mlp_kernel(const float* __restrict__ g,
                           const float* __restrict__ Wl1, const float* __restrict__ bl1,
                           const float* __restrict__ Wl2, const float* __restrict__ bl2,
                           const float* __restrict__ Wl3, const float* __restrict__ bl3,
                           float* __restrict__ out) {
    __shared__ float h1[64];
    __shared__ float h2[32];
    __shared__ float z[2];
    int b = blockIdx.x, t = threadIdx.x;
    const float* gr = g + b * 128;
    if (t < 64) {
        float s = bl1[t];
        for (int k = 0; k < 128; ++k) s += gr[k] * Wl1[k * 64 + t];
        h1[t] = fmaxf(s, 0.f);
    }
    __syncthreads();
    if (t < 32) {
        float s = bl2[t];
        for (int k = 0; k < 64; ++k) s += h1[k] * Wl2[k * 32 + t];
        h2[t] = fmaxf(s, 0.f);
    }
    __syncthreads();
    if (t < 2) {
        float s = bl3[t];
        for (int k = 0; k < 32; ++k) s += h2[k] * Wl3[k * 2 + t];
        z[t] = s;
    }
    __syncthreads();
    if (t < 2) {
        float m = fmaxf(z[0], z[1]);
        float lse = m + logf(expf(z[0] - m) + expf(z[1] - m));
        out[b * 2 + t] = z[t] - lse;
    }
}

// =====================================================================
// host-side orchestration
// =====================================================================
extern "C" void kernel_launch(void* const* d_in, const int* in_sizes, int n_in,
                              void* d_out, int out_size, void* d_ws, size_t ws_size,
                              hipStream_t stream) {
    const float* x     = (const float*)d_in[0];
    const float* eattr = (const float*)d_in[1];
    const float* W1  = (const float*)d_in[2];
    const float* b1  = (const float*)d_in[3];
    const float* Wp1 = (const float*)d_in[4];
    const float* bp1 = (const float*)d_in[5];
    const float* W2  = (const float*)d_in[6];
    const float* b2  = (const float*)d_in[7];
    const float* Wp2 = (const float*)d_in[8];
    const float* bp2 = (const float*)d_in[9];
    const float* W3  = (const float*)d_in[10];
    const float* b3  = (const float*)d_in[11];
    const float* Wp3 = (const float*)d_in[12];
    const float* bp3 = (const float*)d_in[13];
    const float* Wl1 = (const float*)d_in[14];
    const float* bl1 = (const float*)d_in[15];
    const float* Wl2 = (const float*)d_in[16];
    const float* bl2 = (const float*)d_in[17];
    const float* Wl3 = (const float*)d_in[18];
    const float* bl3 = (const float*)d_in[19];
    const int*   eidx = (const int*)d_in[20];
    const int* src = eidx;
    const int* dst = eidx + ETOT;

    char* ws = (char*)d_ws;
    const size_t szBig = (size_t)BN * HF * sizeof(float);    // 32 MB
    float* hW    = (float*)(ws);
    float* hbuf  = (float*)(ws + szBig);
    float* xnb   = (float*)(ws + 2 * szBig);
    float* score = (float*)(ws + 3 * szBig);
    int*   a0    = (int*)  (ws + 3 * szBig + (size_t)BN * 4);
    int*   a1    = (int*)  (ws + 3 * szBig + 2 * (size_t)BN * 4);
    float* gacc  = (float*)(ws + 3 * szBig + 3 * (size_t)BN * 4);

    dim3 gAgg(BG, 2);
    const size_t smemAgg = (size_t)(1024 * 32 + 1024) * sizeof(float);  // 132 KB

    // ---- layer 1 ----
    lin1_kernel<<<(BN * HF + 255) / 256, 256, 0, stream>>>(x, W1, hW);
    gcn_agg_feat<<<gAgg, 256, smemAgg, stream>>>(hW, src, dst, eattr,
                                                 nullptr, 1, b1, 1, hbuf);
    gcn_score<<<BG, 256, 0, stream>>>(hbuf, src, dst, nullptr, Wp1, bp1, score);
    sag_topk<<<BG, 256, 0, stream>>>(score, nullptr, hbuf, 820, a0, xnb);
    readout_kernel<<<BG, 256, 0, stream>>>(xnb, a0, gacc, 0);

    // ---- layer 2 ----
    gemm_h64_wmma<<<BN / 128, 256, 0, stream>>>(xnb, W2, hW);
    gcn_agg_feat<<<gAgg, 256, smemAgg, stream>>>(hW, src, dst, eattr,
                                                 a0, 1, b2, 1, hbuf);
    gcn_score<<<BG, 256, 0, stream>>>(hbuf, src, dst, a0, Wp2, bp2, score);
    sag_topk<<<BG, 256, 0, stream>>>(score, a0, hbuf, 656, a1, xnb);
    readout_kernel<<<BG, 256, 0, stream>>>(xnb, a1, gacc, 1);

    // ---- layer 3 (conv3 has no edge_attr) ----
    gemm_h64_wmma<<<BN / 128, 256, 0, stream>>>(xnb, W3, hW);
    gcn_agg_feat<<<gAgg, 256, smemAgg, stream>>>(hW, src, dst, eattr,
                                                 a1, 0, b3, 1, hbuf);
    gcn_score<<<BG, 256, 0, stream>>>(hbuf, src, dst, a1, Wp3, bp3, score);
    sag_topk<<<BG, 256, 0, stream>>>(score, a1, hbuf, 525, a0, xnb);
    readout_kernel<<<BG, 256, 0, stream>>>(xnb, a0, gacc, 1);

    // ---- MLP head ----
    mlp_kernel<<<BG, 128, 0, stream>>>(gacc, Wl1, bl1, Wl2, bl2, Wl3, bl3,
                                       (float*)d_out);
}
